// TransitionModel_24670292148908
// MI455X (gfx1250) — matrix operand order repaired and checked by reading, not compiled, not run on hardware
//
#include <hip/hip_runtime.h>
#include <hip/hip_bf16.h>

// ---------------------------------------------------------------------------
// Fused Dreamer transition step for MI455X (gfx1250, wave32, WMMA bf16).
//
// Pipeline per 32-row slab (one workgroup, 8 wave32):
//   X = [state|action]            -> hidden = ELU(X @ W_sa + b)        (K=80->96)
//   gi = hidden @ W_ih, gh = h @ W_hh -> GRU gates -> belief           (K=256,N=768 x2)
//   hp = ELU(belief @ W_bp + b)   -> prior = hp @ W_sp  -> mean/std/state
//   hq = ELU([belief|obs] @ W_bq) -> post  = hq @ W_sq  -> mean/std/state (K=1280)
//
// All GEMMs use v_wmma_f32_16x16x32_bf16; intermediates never touch HBM.
// Weights are repacked once per launch (f32 -> bf16, WMMA B-fragment order).
// Activations use branchless native-HW math (v_exp/v_log/v_rcp) to avoid
// EXEC-mask divergence between WMMA clusters.
// ---------------------------------------------------------------------------

typedef __attribute__((ext_vector_type(16))) __bf16 v16bf;
typedef __attribute__((ext_vector_type(8)))  __bf16 v8bf;
typedef __attribute__((ext_vector_type(8)))  float  v8f;

#define DEVINL __device__ __forceinline__

#define BATCH 65536
#define RB    32          // rows per workgroup

// workspace offsets, in bf16 elements (fragment-packed weights, K padded to 32)
#define OFF_SA 0u         // 96x256   -> 48 frags
#define OFF_IH 24576u     // 256x768  -> 384 frags
#define OFF_HH 221184u
#define OFF_BP 417792u    // 256x256  -> 128 frags
#define OFF_SP 483328u    // 256x128  -> 64 frags
#define OFF_BQ 516096u    // 1280x256 -> 640 frags
#define OFF_SQ 843776u
#define WS_ELEMS 876544u  // ~1.67 MB bf16 total

DEVINL __bf16 f2bf(float f) {
  unsigned u = __float_as_uint(f);
  unsigned r = (u + 0x7FFFu + ((u >> 16) & 1u)) >> 16;   // round-to-nearest-even
  unsigned short s = (unsigned short)r;
  __bf16 o; __builtin_memcpy(&o, &s, 2); return o;
}

DEVINL v8f zero8() { v8f z; __builtin_memset(&z, 0, sizeof(z)); return z; }

// ---- branchless activations on native HW transcendentals ----
// elu(x):  exp(min(x,0))-1 <= 0 equals elu for x<=0; fmax picks x when x>0.
DEVINL float eluf(float x) { return fmaxf(x, __expf(fminf(x, 0.f)) - 1.f); }
// sigmoid: 1/(1+e^-x); e^-x -> inf for very negative x, rcp -> 0. Correct limits.
DEVINL float sigm(float x) { return __builtin_amdgcn_rcpf(1.f + __expf(-x)); }
// tanh: 1 - 2/(e^{2x}+1); saturates to +/-1 for large |x| without NaN.
DEVINL float tanh_f(float x) {
  return 1.f - 2.f * __builtin_amdgcn_rcpf(__expf(2.f * x) + 1.f);
}
// softplus, numerically stable and branch-free: max(x,0) + log(1+e^{-|x|}).
DEVINL float softplus_(float x) {
  return fmaxf(x, 0.f) + __logf(1.f + __expf(-fabsf(x)));
}

DEVINL v8f wmma_bf16(v16bf a, v16bf b, v8f c) {
  // (neg_a, A, neg_b, B, c_mod, C, reuse_a, reuse_b)
  return __builtin_amdgcn_wmma_f32_16x16x32_bf16(false, a, false, b,
                                                 (short)0, c, false, false);
}

// A-fragment (16x32 bf16, row-major source in LDS, stride 256 elements).
// ISA layout: lanes 0-15 -> M=0..15; lane group g = lane>>4;
//   elements 0..7  -> K = 8g + e          (VGPR0..3)
//   elements 8..15 -> K = 16 + 8g + (e-8) (VGPR4..7)
// => two contiguous 16-byte runs per lane: two ds_load_b128.
DEVINL v16bf load_a_frag(const __bf16* base, int m0, int kbase, int lane) {
  const __bf16* p = base + (m0 + (lane & 15)) * 256 + kbase + 8 * (lane >> 4);
  v8bf lo = *(const v8bf*)p;
  v8bf hi = *(const v8bf*)(p + 16);
  v16bf r;
#pragma unroll
  for (int i = 0; i < 8; ++i) { r[i] = lo[i]; r[i + 8] = hi[i]; }
  return r;
}

// B-fragment: pre-swizzled by repack kernel; 32 contiguous bytes per lane.
DEVINL v16bf load_b_frag(const __bf16* w, int ktile, int ntile, int Nt, int lane) {
  return *(const v16bf*)(w + (size_t)(((ktile * Nt + ntile) * 32) + lane) * 16);
}

// ---------------------------------------------------------------------------
// Weight repack: f32 row-major [K,N] -> bf16 WMMA B-fragments.
// B-matrix (32x16 bf16) layout: n = lane&15; k = 8*(lane>>4) + e (e<8), 16 + 8*(lane>>4) + (e-8) (e>=8).
// One thread per (fragment, lane); writes 16 contiguous bf16.
// ---------------------------------------------------------------------------
__global__ void repack_w(const float* __restrict__ src, __bf16* __restrict__ dst,
                         int K, int N) {
  int tid = blockIdx.x * blockDim.x + threadIdx.x;
  int Nt = N >> 4;
  int Kt = (K + 31) >> 5;
  int total = Kt * Nt * 32;
  if (tid >= total) return;
  int lane = tid & 31;
  int f    = tid >> 5;
  int kt = f / Nt, nt = f - kt * Nt;
  __bf16* d = dst + (size_t)f * 512 + lane * 16;
  int n  = nt * 16 + (lane & 15);
  int kb = kt * 32 + 8 * (lane >> 4);
#pragma unroll
  for (int e = 0; e < 16; ++e) {
    int k = kb + e + (e >= 8 ? 8 : 0);
    float v = (k < K) ? src[(size_t)k * N + n] : 0.f;
    d[e] = f2bf(v);
  }
}

// ---------------------------------------------------------------------------
// Fused transition kernel.
// ---------------------------------------------------------------------------
__global__ __launch_bounds__(256)
void transition_fused(const float* __restrict__ prev_state,
                      const float* __restrict__ prev_action,
                      const float* __restrict__ prev_belief,
                      const float* __restrict__ observation,
                      const float* __restrict__ eps_prior,
                      const float* __restrict__ eps_post,
                      const float* __restrict__ b_sa, const float* __restrict__ b_ih,
                      const float* __restrict__ b_hh, const float* __restrict__ b_bp,
                      const float* __restrict__ b_sp, const float* __restrict__ b_bq,
                      const float* __restrict__ b_sq,
                      const __bf16* __restrict__ ws,
                      float* __restrict__ out) {
  __shared__ __align__(16) unsigned char smemRaw[49152];
  __bf16* bufA = (__bf16*)(smemRaw);           // X -> hidden -> obs chunks
  __bf16* bufB = (__bf16*)(smemRaw + 16384);   // prev_belief -> belief
  __bf16* bufC = (__bf16*)(smemRaw + 32768);   // hp -> hq
  float*  sMean = (float*)(smemRaw);           // overlays bufA (dead phases)
  float*  sStd  = (float*)(smemRaw + 8192);

  const int tid  = threadIdx.x;
  const int lane = tid & 31;
  const int wave = tid >> 5;
  const int row0 = blockIdx.x * RB;
  const int colL = lane & 15;     // N offset within a tile
  const int rowG = lane >> 4;     // +8 row offset per lane group

  float* out_belief = out;
  float* out_pstate = out + (size_t)BATCH * 256;
  float* out_pmean  = out + (size_t)BATCH * 320;
  float* out_pstd   = out + (size_t)BATCH * 384;
  float* out_qstate = out + (size_t)BATCH * 448;
  float* out_qmean  = out + (size_t)BATCH * 512;
  float* out_qstd   = out + (size_t)BATCH * 576;

  // ---- stage 0: X=[state|action|0pad] (cols 0..95) -> bufA; prev_belief bf16 -> bufB
  for (int idx = tid; idx < RB * 96; idx += 256) {
    int r = idx / 96, c = idx - r * 96;
    int R = row0 + r;
    float v = (c < 64) ? prev_state[(size_t)R * 64 + c]
                       : ((c < 80) ? prev_action[(size_t)R * 16 + (c - 64)] : 0.f);
    bufA[r * 256 + c] = f2bf(v);
  }
  for (int idx = tid; idx < RB * 256; idx += 256) {
    int r = idx >> 8, c = idx & 255;
    bufB[idx] = f2bf(prev_belief[(size_t)(row0 + r) * 256 + c]);
  }
  __syncthreads();

  // ---- stage 1: hidden = ELU(X @ W_sa + b_sa)   [32x256], K=96
  v8f hid[4];
#pragma unroll
  for (int i = 0; i < 4; ++i) {
    int t = wave * 4 + i, mt = t >> 4, nt = t & 15;
    v8f acc = zero8();
    for (int ks = 0; ks < 3; ++ks) {
      v16bf a = load_a_frag(bufA, mt * 16, ks * 32, lane);
      acc = wmma_bf16(a, load_b_frag(ws + OFF_SA, ks, nt, 16, lane), acc);
    }
    float bias = b_sa[nt * 16 + colL];
#pragma unroll
    for (int v = 0; v < 8; ++v) hid[i][v] = eluf(acc[v] + bias);
  }
  __syncthreads();                 // X reads done -> reuse bufA for hidden
#pragma unroll
  for (int i = 0; i < 4; ++i) {
    int t = wave * 4 + i, mt = t >> 4, nt = t & 15;
    int col = nt * 16 + colL;
#pragma unroll
    for (int v = 0; v < 8; ++v)
      bufA[(mt * 16 + v + 8 * rowG) * 256 + col] = f2bf(hid[i][v]);
  }
  __syncthreads();

  // ---- stage 2: GRU -> belief (gi = hidden@W_ih, gh = h@W_hh, torch gate order r,z,n)
  v8f belv[4];
#pragma unroll
  for (int i = 0; i < 4; ++i) {
    int t = wave * 4 + i, mt = t >> 4, nt = t & 15;
    v8f ir = zero8(), iz = zero8(), in_ = zero8();
    v8f hr = zero8(), hz = zero8(), hn  = zero8();
    for (int ks = 0; ks < 8; ++ks) {
      v16bf ah = load_a_frag(bufA, mt * 16, ks * 32, lane);   // hidden
      v16bf ab = load_a_frag(bufB, mt * 16, ks * 32, lane);   // prev_belief
      ir  = wmma_bf16(ah, load_b_frag(ws + OFF_IH, ks, nt,      48, lane), ir);
      iz  = wmma_bf16(ah, load_b_frag(ws + OFF_IH, ks, nt + 16, 48, lane), iz);
      in_ = wmma_bf16(ah, load_b_frag(ws + OFF_IH, ks, nt + 32, 48, lane), in_);
      hr  = wmma_bf16(ab, load_b_frag(ws + OFF_HH, ks, nt,      48, lane), hr);
      hz  = wmma_bf16(ab, load_b_frag(ws + OFF_HH, ks, nt + 16, 48, lane), hz);
      hn  = wmma_bf16(ab, load_b_frag(ws + OFF_HH, ks, nt + 32, 48, lane), hn);
    }
    int col = nt * 16 + colL;
    float bir = b_ih[col], biz = b_ih[256 + col], bin = b_ih[512 + col];
    float bhr = b_hh[col], bhz = b_hh[256 + col], bhn = b_hh[512 + col];
#pragma unroll
    for (int v = 0; v < 8; ++v) {
      int R = row0 + mt * 16 + v + 8 * rowG;
      float rg = sigm(ir[v] + bir + hr[v] + bhr);
      float zg = sigm(iz[v] + biz + hz[v] + bhz);
      float ng = tanh_f(in_[v] + bin + rg * (hn[v] + bhn));
      float h  = prev_belief[(size_t)R * 256 + col];          // f32 residual path
      float bl = (1.f - zg) * ng + zg * h;
      belv[i][v] = bl;
      out_belief[(size_t)R * 256 + col] = bl;
    }
  }
  __syncthreads();                 // prev_belief(bf16) reads done -> belief into bufB
#pragma unroll
  for (int i = 0; i < 4; ++i) {
    int t = wave * 4 + i, mt = t >> 4, nt = t & 15;
    int col = nt * 16 + colL;
#pragma unroll
    for (int v = 0; v < 8; ++v)
      bufB[(mt * 16 + v + 8 * rowG) * 256 + col] = f2bf(belv[i][v]);
  }
  __syncthreads();

  // ---- stage 3: hp = ELU(belief @ W_bp + b_bp) -> bufC
  v8f hpv[4];
#pragma unroll
  for (int i = 0; i < 4; ++i) {
    int t = wave * 4 + i, mt = t >> 4, nt = t & 15;
    v8f acc = zero8();
    for (int ks = 0; ks < 8; ++ks) {
      v16bf a = load_a_frag(bufB, mt * 16, ks * 32, lane);
      acc = wmma_bf16(a, load_b_frag(ws + OFF_BP, ks, nt, 16, lane), acc);
    }
    float bias = b_bp[nt * 16 + colL];
#pragma unroll
    for (int v = 0; v < 8; ++v) hpv[i][v] = eluf(acc[v] + bias);
  }
#pragma unroll
  for (int i = 0; i < 4; ++i) {
    int t = wave * 4 + i, mt = t >> 4, nt = t & 15;
    int col = nt * 16 + colL;
#pragma unroll
    for (int v = 0; v < 8; ++v)
      bufC[(mt * 16 + v + 8 * rowG) * 256 + col] = f2bf(hpv[i][v]);
  }
  __syncthreads();

  // ---- stage 4: prior = hp @ W_sp + b_sp -> mean/std via LDS overlay (bufA is dead)
#pragma unroll
  for (int i = 0; i < 2; ++i) {
    int t = wave * 2 + i, mt = t >> 3, nt = t & 7;
    v8f acc = zero8();
    for (int ks = 0; ks < 8; ++ks) {
      v16bf a = load_a_frag(bufC, mt * 16, ks * 32, lane);
      acc = wmma_bf16(a, load_b_frag(ws + OFF_SP, ks, nt, 8, lane), acc);
    }
    int ncol = nt * 16 + colL;
    float bias = b_sp[ncol];
#pragma unroll
    for (int v = 0; v < 8; ++v) {
      int r = mt * 16 + v + 8 * rowG;
      float val = acc[v] + bias;
      if (ncol < 64) sMean[r * 64 + ncol] = val;
      else           sStd[r * 64 + (ncol - 64)] = softplus_(val) + 1e-5f;
    }
  }
  __syncthreads();
  for (int idx = tid; idx < RB * 64; idx += 256) {
    int r = idx >> 6, c = idx & 63;
    int R = row0 + r;
    float m = sMean[idx], sd = sStd[idx];
    out_pmean[(size_t)R * 64 + c]  = m;
    out_pstd[(size_t)R * 64 + c]   = sd;
    out_pstate[(size_t)R * 64 + c] = m + sd * eps_prior[(size_t)R * 64 + c];
  }
  __syncthreads();

  // ---- stage 5: hq = ELU([belief|obs] @ W_bq + b_bq), K=1280 streamed in chunks
  v8f acc5[4];
#pragma unroll
  for (int i = 0; i < 4; ++i) acc5[i] = zero8();
#pragma unroll
  for (int i = 0; i < 4; ++i) {                       // belief part (K tiles 0..7)
    int t = wave * 4 + i, mt = t >> 4, nt = t & 15;
    for (int ks = 0; ks < 8; ++ks) {
      v16bf a = load_a_frag(bufB, mt * 16, ks * 32, lane);
      acc5[i] = wmma_bf16(a, load_b_frag(ws + OFF_BQ, ks, nt, 16, lane), acc5[i]);
    }
  }
  for (int c = 0; c < 4; ++c) {                       // 4 x 256-col obs chunks
    __syncthreads();
    for (int idx = tid; idx < RB * 256; idx += 256) {
      int r = idx >> 8, cc = idx & 255;
      bufA[idx] = f2bf(observation[(size_t)(row0 + r) * 1024 + c * 256 + cc]);
    }
    __syncthreads();
#pragma unroll
    for (int i = 0; i < 4; ++i) {
      int t = wave * 4 + i, mt = t >> 4, nt = t & 15;
      for (int ks = 0; ks < 8; ++ks) {
        v16bf a = load_a_frag(bufA, mt * 16, ks * 32, lane);
        acc5[i] = wmma_bf16(a, load_b_frag(ws + OFF_BQ, 8 + c * 8 + ks, nt, 16, lane),
                            acc5[i]);
      }
    }
  }
  __syncthreads();
#pragma unroll
  for (int i = 0; i < 4; ++i) {                       // hq -> bufC
    int t = wave * 4 + i, mt = t >> 4, nt = t & 15;
    int col = nt * 16 + colL;
    float bias = b_bq[col];
#pragma unroll
    for (int v = 0; v < 8; ++v)
      bufC[(mt * 16 + v + 8 * rowG) * 256 + col] = f2bf(eluf(acc5[i][v] + bias));
  }
  __syncthreads();

  // ---- stage 6: posterior = hq @ W_sq + b_sq -> mean/std/state
#pragma unroll
  for (int i = 0; i < 2; ++i) {
    int t = wave * 2 + i, mt = t >> 3, nt = t & 7;
    v8f acc = zero8();
    for (int ks = 0; ks < 8; ++ks) {
      v16bf a = load_a_frag(bufC, mt * 16, ks * 32, lane);
      acc = wmma_bf16(a, load_b_frag(ws + OFF_SQ, ks, nt, 8, lane), acc);
    }
    int ncol = nt * 16 + colL;
    float bias = b_sq[ncol];
#pragma unroll
    for (int v = 0; v < 8; ++v) {
      int r = mt * 16 + v + 8 * rowG;
      float val = acc[v] + bias;
      if (ncol < 64) sMean[r * 64 + ncol] = val;
      else           sStd[r * 64 + (ncol - 64)] = softplus_(val) + 1e-5f;
    }
  }
  __syncthreads();
  for (int idx = tid; idx < RB * 64; idx += 256) {
    int r = idx >> 6, c = idx & 63;
    int R = row0 + r;
    float m = sMean[idx], sd = sStd[idx];
    out_qmean[(size_t)R * 64 + c]  = m;
    out_qstd[(size_t)R * 64 + c]   = sd;
    out_qstate[(size_t)R * 64 + c] = m + sd * eps_post[(size_t)R * 64 + c];
  }
}

// ---------------------------------------------------------------------------
extern "C" void kernel_launch(void* const* d_in, const int* in_sizes, int n_in,
                              void* d_out, int out_size, void* d_ws, size_t ws_size,
                              hipStream_t stream) {
  (void)in_sizes; (void)n_in; (void)out_size; (void)ws_size;

  const float* prev_state  = (const float*)d_in[0];
  const float* prev_action = (const float*)d_in[1];
  const float* prev_belief = (const float*)d_in[2];
  const float* observation = (const float*)d_in[3];
  const float* eps_prior   = (const float*)d_in[4];
  const float* eps_post    = (const float*)d_in[5];
  const float* b_sa = (const float*)d_in[7];
  const float* b_ih = (const float*)d_in[9];
  const float* b_hh = (const float*)d_in[11];
  const float* b_bp = (const float*)d_in[13];
  const float* b_sp = (const float*)d_in[15];
  const float* b_bq = (const float*)d_in[17];
  const float* b_sq = (const float*)d_in[19];

  // Repack all 7 weight matrices into bf16 WMMA B-fragment order in d_ws
  // (needs WS_ELEMS*2 ~= 1.68 MB of workspace; L2-resident thereafter).
  auto rp = [&](const void* w, unsigned off_elems, int K, int N) {
    __bf16* dst = (__bf16*)((char*)d_ws + (size_t)off_elems * 2);
    int Kt = (K + 31) / 32, Nt = N / 16;
    int total = Kt * Nt * 32;
    repack_w<<<(total + 255) / 256, 256, 0, stream>>>((const float*)w, dst, K, N);
  };
  rp(d_in[6],  OFF_SA,   80,  256);
  rp(d_in[8],  OFF_IH,  256,  768);
  rp(d_in[10], OFF_HH,  256,  768);
  rp(d_in[12], OFF_BP,  256,  256);
  rp(d_in[14], OFF_SP,  256,  128);
  rp(d_in[16], OFF_BQ, 1280,  256);
  rp(d_in[18], OFF_SQ,  256,  128);

  transition_fused<<<BATCH / RB, 256, 0, stream>>>(
      prev_state, prev_action, prev_belief, observation, eps_prior, eps_post,
      b_sa, b_ih, b_hh, b_bp, b_sp, b_bq, b_sq,
      (const __bf16*)d_ws, (float*)d_out);
}
